// MultiHeadSelfAttention_3839700762768
// MI455X (gfx1250) — matrix operand order repaired and checked by reading, not compiled, not run on hardware
//
#include <hip/hip_runtime.h>

// ---------------------------------------------------------------------------
// MHA forward for MI455X (gfx1250).
// All matmuls: bf16 WMMA, f32 accumulate. All LDS staging in the hot loops is
// pure async byte-copy (global_load_async_to_lds_b128) with double buffering.
// Pipeline:
//   prep   : xb = bf16(x); Wt_* = bf16(W_*^T)  (LDS-tiled transpose)
//   gemm<1>: Q = xb Wt_q + b -> bf16 [B,H,S,Dh]
//   gemm<2>: K = xb Wt_k + b -> bf16 [B,H,Dh,S]  (pre-transposed per head)
//   gemm<1>: V = xb Wt_v + b -> bf16 [B,H,S,Dh]
//   attn   : flash attention (double-buffered async K/V tiles) -> bf16 ctx
//   gemm<0>: out = ctx Wt_o + b -> fp32
// ---------------------------------------------------------------------------

typedef __bf16 bf16;
typedef __attribute__((ext_vector_type(16))) __bf16 v16bf;
typedef __attribute__((ext_vector_type(8)))  __bf16 v8bf;
typedef __attribute__((ext_vector_type(4)))  __bf16 v4bf;
typedef __attribute__((ext_vector_type(8)))  float   v8f;

#define D_MODEL 1024
#define SEQ     2048
#define BATCH   2
#define NHEAD   16
#define HDIM    64
#define MTOT    (BATCH * SEQ)   // 4096

__device__ __forceinline__ bf16 f2bf(float f) { return (bf16)f; }

__device__ __forceinline__ unsigned lds_off(const void* p) {
  return (unsigned)(unsigned long long)(uintptr_t)p;
}
__device__ __forceinline__ void async_g2l_b128(unsigned lds_addr, const bf16* g) {
  asm volatile("global_load_async_to_lds_b128 %0, %1, off"
               :: "v"(lds_addr), "v"(g) : "memory");
}
__device__ __forceinline__ void wait_async0() {
  asm volatile("s_wait_asynccnt 0x0" ::: "memory");
}

// A-fragment (16x32 bf16, MxK), ISA 7.12.2 layout
__device__ __forceinline__ v16bf frag_a(const bf16* base, int ld, int lane) {
  const int lm = lane & 15, lh = lane >> 4;
  const bf16* row = base + lm * ld;
  v8bf x0 = *reinterpret_cast<const v8bf*>(row + lh * 8);
  v8bf x1 = *reinterpret_cast<const v8bf*>(row + 16 + lh * 8);
  v16bf r;
#pragma unroll
  for (int i = 0; i < 8; ++i) { r[i] = x0[i]; r[i + 8] = x1[i]; }
  return r;
}
// B-fragment (32x16 bf16, KxN): lane holds column n = lane%16, k in table order
__device__ __forceinline__ v16bf frag_b(const bf16* base, int ld, int lane) {
  const int ln = lane & 15, lh = lane >> 4;
  const bf16* col = base + ln;
  v16bf r;
#pragma unroll
  for (int i = 0; i < 8; ++i) { r[i] = col[(lh * 8 + i) * ld]; r[i + 8] = col[(16 + lh * 8 + i) * ld]; }
  return r;
}

// ---------------------------------------------------------------------------
// prep: elementwise fp32 -> bf16 (x and ctx feed GEMMs as bf16)
// ---------------------------------------------------------------------------
__global__ __launch_bounds__(256) void cvt_f32_bf16(
    const float* __restrict__ src, bf16* __restrict__ dst) {
  const size_t i = (size_t)blockIdx.x * 256 + threadIdx.x;   // float4 index
  float4 v = reinterpret_cast<const float4*>(src)[i];
  v4bf o = { f2bf(v.x), f2bf(v.y), f2bf(v.z), f2bf(v.w) };
  reinterpret_cast<v4bf*>(dst)[i] = o;
}

// ---------------------------------------------------------------------------
// prep: Wt[k][n] = bf16(W[n][k]), 64x64 LDS-tiled transpose, coalesced both ways
// ---------------------------------------------------------------------------
__global__ __launch_bounds__(256) void transpose_w_bf16(
    const float* __restrict__ W, bf16* __restrict__ Wt) {
  __shared__ __align__(16) bf16 lsT[64 * 72];   // [n_local][k_local]
  const int tid = threadIdx.x;
  const int k0 = blockIdx.x * 64, n0 = blockIdx.y * 64;
  const int r0 = tid >> 4;            // 0..15
  const int c  = (tid & 15) * 4;      // k_local
#pragma unroll
  for (int rr = 0; rr < 64; rr += 16) {
    const int n = r0 + rr;
    float4 w = *reinterpret_cast<const float4*>(W + (size_t)(n0 + n) * D_MODEL + k0 + c);
    bf16* d = lsT + n * 72 + c;
    d[0] = f2bf(w.x); d[1] = f2bf(w.y); d[2] = f2bf(w.z); d[3] = f2bf(w.w);
  }
  __syncthreads();
#pragma unroll
  for (int j = 0; j < 2; ++j) {
    const int chunk = tid + j * 256;          // 512 chunks
    const int k = chunk >> 3, cc = (chunk & 7) * 8;
    v8bf o;
#pragma unroll
    for (int i = 0; i < 8; ++i) o[i] = lsT[(cc + i) * 72 + k];
    *reinterpret_cast<v8bf*>(Wt + (size_t)(k0 + k) * D_MODEL + n0 + cc) = o;
  }
}

// ---------------------------------------------------------------------------
// C[4096,1024] = A[4096,1024](bf16) * Wt[1024,1024](bf16,[K][N]) + bias
// MODE 0: fp32 [M,N] | MODE 1: bf16 [B,H,S,Dh] | MODE 2: bf16 [B,H,Dh,S]
// 256 threads (8 waves), tile 128x128, K-slice 32, double-buffered async LDS.
// Wave (wm,wn) owns 32x64 -> 8 WMMA per slice. One barrier per slice.
// ---------------------------------------------------------------------------
template <int MODE>
__global__ __launch_bounds__(256) void gemm_bias_wmma(
    const bf16* __restrict__ A, const bf16* __restrict__ Wt,
    const float* __restrict__ bias, void* __restrict__ OutP) {
  __shared__ __align__(16) bf16 lsA[2][128 * 40];   // [row][k] stride 40
  __shared__ __align__(16) bf16 lsB[2][32 * 136];   // [k][n]   stride 136

  float* __restrict__ OutF = (float*)OutP;
  bf16*  __restrict__ OutB = (bf16*)OutP;

  const int tid  = threadIdx.x;
  const int lane = tid & 31, wave = tid >> 5;
  const int wm = wave & 3, wn = wave >> 2;
  const int n0 = blockIdx.x * 128;
  const int m0 = blockIdx.y * 128;

  v8f vzero = {};
  v8f acc[2][4];
#pragma unroll
  for (int i = 0; i < 2; ++i)
#pragma unroll
    for (int t = 0; t < 4; ++t) acc[i][t] = vzero;

  // async stage of one K-slice: 2 A-chunks + 2 B-chunks per thread (16B each)
  auto stage = [&](int buf, int kb) {
#pragma unroll
    for (int j = 0; j < 2; ++j) {
      const int ca = tid + j * 256;                 // A: 128 rows x 4 chunks
      const int ra = ca >> 2, oa = (ca & 3) * 8;
      async_g2l_b128(lds_off(&lsA[buf][ra * 40 + oa]),
                     A + (size_t)(m0 + ra) * D_MODEL + kb + oa);
      const int cb = tid + j * 256;                 // B: 32 rows x 16 chunks
      const int rb = cb >> 4, ob = (cb & 15) * 8;
      async_g2l_b128(lds_off(&lsB[buf][rb * 136 + ob]),
                     Wt + (size_t)(kb + rb) * D_MODEL + n0 + ob);
    }
  };

  stage(0, 0);
  int buf = 0;
  for (int kb = 0; kb < D_MODEL; kb += 32) {
    wait_async0();        // own copies for this slice landed
    __syncthreads();      // everyone's landed; other buffer free for reuse
    if (kb + 32 < D_MODEL) stage(buf ^ 1, kb + 32);   // overlap with compute

    v16bf af[2];
#pragma unroll
    for (int i = 0; i < 2; ++i)
      af[i] = frag_a(&lsA[buf][(wm * 32 + i * 16) * 40], 40, lane);
#pragma unroll
    for (int t = 0; t < 4; ++t) {
      v16bf bfr = frag_b(&lsB[buf][wn * 64 + t * 16], 136, lane);
#pragma unroll
      for (int i = 0; i < 2; ++i)
        acc[i][t] = __builtin_amdgcn_wmma_f32_16x16x32_bf16(
            false, af[i], false, bfr, (short)0, acc[i][t], false, false);
    }
    buf ^= 1;
  }

  const int lm = lane & 15, lh = lane >> 4;
#pragma unroll
  for (int i = 0; i < 2; ++i)
#pragma unroll
    for (int t = 0; t < 4; ++t)
#pragma unroll
      for (int r = 0; r < 8; ++r) {
        const int m = m0 + wm * 32 + i * 16 + r + 8 * lh;
        const int n = n0 + wn * 64 + t * 16 + lm;
        const float v = acc[i][t][r] + bias[n];
        if (MODE == 0) {
          OutF[(size_t)m * D_MODEL + n] = v;
        } else {
          const int b = m >> 11, s = m & (SEQ - 1);
          const int h = n >> 6,  d = n & (HDIM - 1);
          if (MODE == 1)
            OutB[(((size_t)(b * NHEAD + h)) * SEQ + s) * HDIM + d] = f2bf(v);
          else
            OutB[(((size_t)(b * NHEAD + h)) * HDIM + d) * SEQ + s] = f2bf(v);
        }
      }
}

// ---------------------------------------------------------------------------
// Flash attention. Q,V bf16 [B,H,S,Dh]; Kt bf16 [B,H,Dh,S]; Ctx bf16 [B,S,D].
// 128 threads (4 waves), 64 q-rows/block, 64-key tiles, double-buffered async.
// ---------------------------------------------------------------------------
__global__ __launch_bounds__(128) void attn_wmma(
    const bf16* __restrict__ Q, const bf16* __restrict__ Kt,
    const bf16* __restrict__ V, bf16* __restrict__ Ctx) {
  __shared__ __align__(16) bf16 lsQ [64 * 72];      // [q][d]
  __shared__ __align__(16) bf16 lsKT[2][64 * 72];   // [d][key]
  __shared__ __align__(16) bf16 lsV [2][64 * 72];   // [key][d]
  __shared__ __align__(16) bf16 lsP [64 * 72];      // [q][key]

  const int tid  = threadIdx.x;
  const int lane = tid & 31, wave = tid >> 5;
  const int lm = lane & 15,  lh = lane >> 4;
  const int q0 = blockIdx.x * 64;
  const int bh = blockIdx.y;
  const size_t head_sd = (size_t)bh * SEQ * HDIM;   // [s][d] (Q, V)
  const size_t head_ds = (size_t)bh * HDIM * SEQ;   // [d][s] (Kt)

  auto stage_kv = [&](int buf, int k0) {
#pragma unroll
    for (int j = 0; j < 4; ++j) {
      const int chunk = tid + j * 128;              // 512 chunks per tile
      const int r = chunk >> 3, c = (chunk & 7) * 8;
      async_g2l_b128(lds_off(&lsKT[buf][r * 72 + c]),
                     Kt + head_ds + (size_t)r * SEQ + k0 + c);
      async_g2l_b128(lds_off(&lsV[buf][r * 72 + c]),
                     V + head_sd + (size_t)(k0 + r) * HDIM + c);
    }
  };

  // stage Q (once) + first K/V tile
#pragma unroll
  for (int j = 0; j < 4; ++j) {
    const int chunk = tid + j * 128;
    const int r = chunk >> 3, c = (chunk & 7) * 8;
    async_g2l_b128(lds_off(lsQ + r * 72 + c),
                   Q + head_sd + (size_t)(q0 + r) * HDIM + c);
  }
  stage_kv(0, 0);

  float mrun[8], lrun[8];
  v8f vzero = {};
  v8f oacc[4];
#pragma unroll
  for (int r = 0; r < 8; ++r) { mrun[r] = -3.0e38f; lrun[r] = 0.0f; }
#pragma unroll
  for (int t = 0; t < 4; ++t) oacc[t] = vzero;

  int buf = 0;
  for (int k0 = 0; k0 < SEQ; k0 += 64) {
    wait_async0();
    __syncthreads();
    if (k0 + 64 < SEQ) stage_kv(buf ^ 1, k0 + 64);   // overlap with compute

    // ---- S = Q K^T : 16 q-rows x 64 keys per wave, Dh = 64 = 2x32
    v16bf aq0 = frag_a(lsQ + wave * 16 * 72, 72, lane);
    v16bf aq1 = frag_a(lsQ + wave * 16 * 72 + 32, 72, lane);
    v8f sc[4];
#pragma unroll
    for (int t = 0; t < 4; ++t) {
      sc[t] = vzero;
      sc[t] = __builtin_amdgcn_wmma_f32_16x16x32_bf16(
          false, aq0, false, frag_b(&lsKT[buf][t * 16], 72, lane), (short)0, sc[t], false, false);
      sc[t] = __builtin_amdgcn_wmma_f32_16x16x32_bf16(
          false, aq1, false, frag_b(&lsKT[buf][32 * 72 + t * 16], 72, lane), (short)0, sc[t], false, false);
    }

    // ---- online softmax (scale = 1/sqrt(64))
#pragma unroll
    for (int t = 0; t < 4; ++t)
#pragma unroll
      for (int r = 0; r < 8; ++r) sc[t][r] *= 0.125f;

#pragma unroll
    for (int r = 0; r < 8; ++r) {
      float vmax = fmaxf(fmaxf(sc[0][r], sc[1][r]), fmaxf(sc[2][r], sc[3][r]));
      vmax = fmaxf(vmax, __shfl_xor(vmax, 1));
      vmax = fmaxf(vmax, __shfl_xor(vmax, 2));
      vmax = fmaxf(vmax, __shfl_xor(vmax, 4));
      vmax = fmaxf(vmax, __shfl_xor(vmax, 8));   // within 16-lane half
      const float mnew = fmaxf(mrun[r], vmax);
      const float corr = __expf(mrun[r] - mnew);
      mrun[r] = mnew;
      float rsum = 0.0f;
#pragma unroll
      for (int t = 0; t < 4; ++t) {
        const float p = __expf(sc[t][r] - mnew);
        sc[t][r] = p;
        rsum += p;
      }
      rsum += __shfl_xor(rsum, 1);
      rsum += __shfl_xor(rsum, 2);
      rsum += __shfl_xor(rsum, 4);
      rsum += __shfl_xor(rsum, 8);
      lrun[r] = lrun[r] * corr + rsum;
#pragma unroll
      for (int t = 0; t < 4; ++t) oacc[t][r] *= corr;
    }

    // ---- P (C-layout) -> LDS bf16 to re-read in A-layout
#pragma unroll
    for (int t = 0; t < 4; ++t)
#pragma unroll
      for (int r = 0; r < 8; ++r)
        lsP[(wave * 16 + r + 8 * lh) * 72 + t * 16 + lm] = f2bf(sc[t][r]);
    __syncthreads();

    // ---- O += P V  (K-dim = 64 keys = 2x32)
    v16bf ap0 = frag_a(lsP + wave * 16 * 72, 72, lane);
    v16bf ap1 = frag_a(lsP + wave * 16 * 72 + 32, 72, lane);
#pragma unroll
    for (int t = 0; t < 4; ++t) {
      oacc[t] = __builtin_amdgcn_wmma_f32_16x16x32_bf16(
          false, ap0, false, frag_b(&lsV[buf][t * 16], 72, lane), (short)0, oacc[t], false, false);
      oacc[t] = __builtin_amdgcn_wmma_f32_16x16x32_bf16(
          false, ap1, false, frag_b(&lsV[buf][32 * 72 + t * 16], 72, lane), (short)0, oacc[t], false, false);
    }
    buf ^= 1;
  }

  // ---- normalize, write ctx as bf16 [B, S, D_MODEL]
  const int b = bh >> 4, h = bh & 15;
#pragma unroll
  for (int t = 0; t < 4; ++t)
#pragma unroll
    for (int r = 0; r < 8; ++r) {
      const int s = q0 + wave * 16 + r + 8 * lh;
      const int d = t * 16 + lm;
      Ctx[((size_t)(b * SEQ + s)) * D_MODEL + h * HDIM + d] = f2bf(oacc[t][r] / lrun[r]);
    }
}

// ---------------------------------------------------------------------------
extern "C" void kernel_launch(void* const* d_in, const int* in_sizes, int n_in,
                              void* d_out, int out_size, void* d_ws, size_t ws_size,
                              hipStream_t stream) {
  const float* x  = (const float*)d_in[0];
  const float* Wq = (const float*)d_in[1]; const float* bq = (const float*)d_in[2];
  const float* Wk = (const float*)d_in[3]; const float* bk = (const float*)d_in[4];
  const float* Wv = (const float*)d_in[5]; const float* bv = (const float*)d_in[6];
  const float* Wo = (const float*)d_in[7]; const float* bo = (const float*)d_in[8];
  float* out = (float*)d_out;

  // Workspace (bf16 unless noted):
  //  xb[4096x1024] | Wtq|Wtk|Wtv|Wto[1024x1024] | Q|Kt|V[4096x1024] | Ctx[4096x1024]
  bf16* xb  = (bf16*)d_ws;
  bf16* Wtq = xb  + (size_t)MTOT * D_MODEL;
  bf16* Wtk = Wtq + (size_t)D_MODEL * D_MODEL;
  bf16* Wtv = Wtk + (size_t)D_MODEL * D_MODEL;
  bf16* Wto = Wtv + (size_t)D_MODEL * D_MODEL;
  bf16* Qb  = Wto + (size_t)D_MODEL * D_MODEL;
  bf16* Ktb = Qb  + (size_t)MTOT * D_MODEL;
  bf16* Vb  = Ktb + (size_t)MTOT * D_MODEL;
  bf16* Ctx = Vb  + (size_t)MTOT * D_MODEL;

  // prep: conversions + weight transposes
  cvt_f32_bf16<<<dim3((MTOT * D_MODEL / 4) / 256), dim3(256), 0, stream>>>(x, xb);
  dim3 tgrid(D_MODEL / 64, D_MODEL / 64);
  transpose_w_bf16<<<tgrid, dim3(256), 0, stream>>>(Wq, Wtq);
  transpose_w_bf16<<<tgrid, dim3(256), 0, stream>>>(Wk, Wtk);
  transpose_w_bf16<<<tgrid, dim3(256), 0, stream>>>(Wv, Wtv);
  transpose_w_bf16<<<tgrid, dim3(256), 0, stream>>>(Wo, Wto);

  dim3 gblk(256);
  dim3 ggrid(D_MODEL / 128, MTOT / 128);   // 8 x 32
  gemm_bias_wmma<1><<<ggrid, gblk, 0, stream>>>(xb, Wtq, bq, Qb);
  gemm_bias_wmma<2><<<ggrid, gblk, 0, stream>>>(xb, Wtk, bk, Ktb);
  gemm_bias_wmma<1><<<ggrid, gblk, 0, stream>>>(xb, Wtv, bv, Vb);

  dim3 agrid(SEQ / 64, BATCH * NHEAD);     // 32 x 32
  attn_wmma<<<agrid, dim3(128), 0, stream>>>(Qb, Ktb, Vb, Ctx);

  gemm_bias_wmma<0><<<ggrid, gblk, 0, stream>>>(Ctx, Wto, bo, out);
}
